// VectorQuantizer_86775519248430
// MI455X (gfx1250) — matrix operand order repaired and checked by reading, not compiled, not run on hardware
//
#include <hip/hip_runtime.h>
#include <hip/hip_bf16.h>
#include <math.h>

// ---------------------------------------------------------------------------
// VQ-VAE vector quantizer for MI455X (gfx1250, wave32, WMMA).
//
// distances = ||x||^2 - 2 x@w + ||w||^2 ; argmin over K ; one-hot ; gather ;
// loss = 1.25 * mean((q-x)^2) ; perplexity from counts.
//
// GEMM path: bf16x3 split (hi+lo) through v_wmma_f32_16x16x32_bf16 to recover
// ~fp32 argmin fidelity at bf16 matrix-core rates. Distances are never
// materialized: each wave keeps a running per-row (min,idx) while sweeping K.
// B fragments are software-pipelined one tile ahead so each load pair has a
// full d-tile of WMMA latency for cover (occupancy is VGPR-limited).
// ---------------------------------------------------------------------------

typedef __attribute__((ext_vector_type(16))) __bf16 v16bf;
typedef __attribute__((ext_vector_type(8)))  float  v8f;
typedef __attribute__((ext_vector_type(4)))  float  v4f;
typedef __attribute__((ext_vector_type(2)))  float  v2f;

namespace {
constexpr int kN = 32768;        // B*H*W rows
constexpr int kD = 256;          // embedding dim
constexpr int kK = 8192;         // codebook entries
constexpr int kNKT = kK / 16;    // 512 k-tiles
constexpr int kNDT = kD / 32;    // 8 depth-tiles
constexpr int kNRT = kN / 16;    // 2048 row-tiles

// workspace byte offsets (total ~51 MB; each region 32B-aligned)
constexpr size_t OFF_WHI  = 0;                                        // 4 MiB
constexpr size_t OFF_WLO  = OFF_WHI + (size_t)kNKT * kNDT * 512 * 2;  // 4 MiB
constexpr size_t OFF_XHI  = OFF_WLO + (size_t)kNKT * kNDT * 512 * 2;  // 16 MiB
constexpr size_t OFF_XLO  = OFF_XHI + (size_t)kNRT * kNDT * 512 * 2;  // 16 MiB
constexpr size_t OFF_WT   = OFF_XLO + (size_t)kNRT * kNDT * 512 * 2;  // 8 MiB (w transposed)
constexpr size_t OFF_WSQ  = OFF_WT  + (size_t)kK * kD * 4;            // 32 KiB
constexpr size_t OFF_XSQ  = OFF_WSQ + (size_t)kK * 4;                 // 128 KiB
constexpr size_t OFF_IDX  = OFF_XSQ + (size_t)kN * 4;                 // 128 KiB (int idx)
constexpr size_t OFF_CNT  = OFF_IDX + (size_t)kN * 4;                 // 32 KiB (counts)
constexpr size_t OFF_LACC = OFF_CNT + (size_t)kK * 4;                 // 4 B (loss accum)

// output float offsets (tuple concatenated flat in return order)
constexpr size_t OUT_Q    = 0;                        // quantized_st [N,D]
constexpr size_t OUT_LOSS = (size_t)kN * kD;          // scalar
constexpr size_t OUT_PPL  = OUT_LOSS + 1;             // scalar
constexpr size_t OUT_ENC  = OUT_PPL + 1;              // encodings [N,K]
constexpr size_t OUT_IDX  = OUT_ENC + (size_t)kN * kK;// encoding_indices [N]
} // namespace

__device__ __forceinline__ unsigned short f2bf_rne(float f) {
  unsigned int u = __float_as_uint(f);
  unsigned int r = u + 0x7FFFu + ((u >> 16) & 1u);   // round-to-nearest-even
  return (unsigned short)(r >> 16);
}
__device__ __forceinline__ float bf2f(unsigned short s) {
  return __uint_as_float(((unsigned int)s) << 16);
}

// --- zero counts[K] + loss accumulator (harness does not re-poison) --------
__global__ void vq_zero_kernel(unsigned int* __restrict__ cnt) {
  int i = blockIdx.x * 256 + threadIdx.x;
  if (i < kK + 1) cnt[i] = 0u;   // counts followed directly by loss accum
}

// --- ||x||^2 per row --------------------------------------------------------
__global__ void vq_xsq_kernel(const float* __restrict__ x, float* __restrict__ xsq) {
  int row = blockIdx.x * 256 + threadIdx.x;
  const v4f* p = reinterpret_cast<const v4f*>(x + (size_t)row * kD);
  float s = 0.f;
#pragma unroll 8
  for (int i = 0; i < kD / 4; ++i) {
    v4f v = p[i];
    s += v.x * v.x + v.y * v.y + v.z * v.z + v.w * v.w;
  }
  xsq[row] = s;
}

// --- ||w||^2 per code + w transposed [K,D] for coalesced gather ------------
__global__ void vq_wsq_wt_kernel(const float* __restrict__ w,
                                 float* __restrict__ wsq, float* __restrict__ wt) {
  int k = blockIdx.x * 256 + threadIdx.x;
  float s = 0.f;
  for (int d = 0; d < kD; ++d) {
    float v = w[(size_t)d * kK + k];        // coalesced across k
    wt[(size_t)k * kD + d] = v;
    s += v * v;
  }
  wsq[k] = s;
}

// --- pack x into WMMA A-matrix 16x32 bf16 lane layout (hi + lo split) ------
// A layout (ISA 7.12.2): lane m(0-15) row m; VGPR v pair = K {2v,2v+1}+half*8
// for v<4, and 16+{...} for v>=4; lanes 16-31 shift K by 8.
__global__ void vq_packx_kernel(const float* __restrict__ x,
                                unsigned short* __restrict__ xhi,
                                unsigned short* __restrict__ xlo) {
  int wave = threadIdx.x >> 5, lane = threadIdx.x & 31;
  int tile = blockIdx.x * 8 + wave;       // tile = rt*kNDT + dt
  int rt = tile >> 3, dt = tile & 7;
  int m = lane & 15, half = lane >> 4;
  size_t base = ((size_t)tile * 32 + lane) * 16;
  const float* xr = x + (size_t)(rt * 16 + m) * kD + dt * 32;
#pragma unroll
  for (int e = 0; e < 16; ++e) {
    int v = e >> 1, pos = e & 1;
    int kd = ((v < 4) ? 0 : 16) + half * 8 + (v & 3) * 2 + pos;
    float f = xr[kd];
    unsigned short h = f2bf_rne(f);
    xhi[base + e] = h;
    xlo[base + e] = f2bf_rne(f - bf2f(h));
  }
}

// --- pack w into WMMA B-matrix 32x16 bf16 lane layout (hi + lo split) ------
// B layout: lane n(0-15) column n with K=e for e<8, K=16+(e-8) for e>=8;
// lanes 16-31 shift K by 8 for the same columns (matches A).
__global__ void vq_packw_kernel(const float* __restrict__ w,
                                unsigned short* __restrict__ whi,
                                unsigned short* __restrict__ wlo) {
  int wave = threadIdx.x >> 5, lane = threadIdx.x & 31;
  int tile = blockIdx.x * 8 + wave;       // tile = kt*kNDT + dt
  int kt = tile >> 3, dt = tile & 7;
  int n = lane & 15, half = lane >> 4;
  size_t base = ((size_t)tile * 32 + lane) * 16;
  int k = kt * 16 + n;
#pragma unroll
  for (int e = 0; e < 16; ++e) {
    int kd = ((e < 8) ? 0 : 16) + half * 8 + (e & 7);
    float f = w[(size_t)(dt * 32 + kd) * kK + k];
    unsigned short h = f2bf_rne(f);
    whi[base + e] = h;
    wlo[base + e] = f2bf_rne(f - bf2f(h));
  }
}

// --- main: bf16x3 WMMA distance GEMM + fused running argmin ----------------
// One wave owns a 16-row tile; A-fragments (hi+lo, 8 d-tiles) stay resident
// in VGPRs; we sweep all 512 k-tiles. C/D layout: slot r <-> row
// rowbase + half*8 + r, lane's column = kt*16 + (lane&15). B fragments are
// prefetched one (kt,dt) tile ahead so the s_wait for the consuming WMMA
// skips the freshly-issued loads.
__global__ void __launch_bounds__(128)
vq_gemm_argmin_kernel(const unsigned short* __restrict__ xhi,
                      const unsigned short* __restrict__ xlo,
                      const unsigned short* __restrict__ whi,
                      const unsigned short* __restrict__ wlo,
                      const float* __restrict__ xsq,
                      const float* __restrict__ wsq,
                      int* __restrict__ idx,
                      unsigned int* __restrict__ cnt,
                      float* __restrict__ out_idx) {
  int wave = threadIdx.x >> 5, lane = threadIdx.x & 31;
  int rt = blockIdx.x * 4 + wave;
  int n = lane & 15, half = lane >> 4;
  int rowbase = rt * 16;

  v16bf ahi[kNDT], alo[kNDT];
#pragma unroll
  for (int dt = 0; dt < kNDT; ++dt) {
    size_t off = ((size_t)(rt * kNDT + dt) * 32 + lane) * 16;
    ahi[dt] = *reinterpret_cast<const v16bf*>(xhi + off);
    alo[dt] = *reinterpret_cast<const v16bf*>(xlo + off);
  }
  float xs[8];
#pragma unroll
  for (int r = 0; r < 8; ++r) xs[r] = xsq[rowbase + half * 8 + r];

  float minv[8];
  int mini[8];
#pragma unroll
  for (int r = 0; r < 8; ++r) { minv[r] = 3.4e38f; mini[r] = 0; }

  // software pipeline: preload tile 0, then load t+1 while computing t
  constexpr int kNT = kNKT * kNDT;                   // 4096 (kt,dt) tiles
  size_t lane16 = (size_t)lane * 16;
  v16bf bh_cur = *reinterpret_cast<const v16bf*>(whi + lane16);
  v16bf bl_cur = *reinterpret_cast<const v16bf*>(wlo + lane16);

  for (int kt = 0; kt < kNKT; ++kt) {
    v8f acc = {0.f, 0.f, 0.f, 0.f, 0.f, 0.f, 0.f, 0.f};
#pragma unroll
    for (int dt = 0; dt < kNDT; ++dt) {
      int t = kt * kNDT + dt;
      int tn = (t + 1 < kNT) ? (t + 1) : t;
      size_t noff = (size_t)tn * 512 + lane16;       // 512 elems per tile
      v16bf bh_nxt = *reinterpret_cast<const v16bf*>(whi + noff);
      v16bf bl_nxt = *reinterpret_cast<const v16bf*>(wlo + noff);
      // hi*hi + hi*lo + lo*hi  (drop lo*lo): ~fp32-accurate dot product
      acc = __builtin_amdgcn_wmma_f32_16x16x32_bf16(false, ahi[dt], false, bh_cur,
                                                    (short)0, acc, false, false);
      acc = __builtin_amdgcn_wmma_f32_16x16x32_bf16(false, ahi[dt], false, bl_cur,
                                                    (short)0, acc, false, false);
      acc = __builtin_amdgcn_wmma_f32_16x16x32_bf16(false, alo[dt], false, bh_cur,
                                                    (short)0, acc, false, false);
      bh_cur = bh_nxt;
      bl_cur = bl_nxt;
    }
    float wsv = wsq[kt * 16 + n];
    int kidx = kt * 16 + n;
#pragma unroll
    for (int r = 0; r < 8; ++r) {
      float dist = xs[r] - 2.f * acc[r] + wsv;
      if (dist < minv[r]) { minv[r] = dist; mini[r] = kidx; }
    }
  }

  // argmin across the 16 lanes holding one row (xor masks 1,2,4,8 stay
  // inside each 16-lane half of the wave32). Ties -> lowest index (jnp.argmin).
#pragma unroll
  for (int mask = 1; mask < 16; mask <<= 1) {
#pragma unroll
    for (int r = 0; r < 8; ++r) {
      float ov = __shfl_xor(minv[r], mask, 32);
      int oi = __shfl_xor(mini[r], mask, 32);
      if (ov < minv[r] || (ov == minv[r] && oi < mini[r])) {
        minv[r] = ov; mini[r] = oi;
      }
    }
  }
  if (n == 0) {   // lanes 0 (rows 0-7) and 16 (rows 8-15)
#pragma unroll
    for (int r = 0; r < 8; ++r) {
      int row = rowbase + half * 8 + r;
      idx[row] = mini[r];
      out_idx[row] = (float)mini[r];
      atomicAdd(&cnt[mini[r]], 1u);
    }
  }
}

// --- gather quantized rows (coalesced via w^T) + loss sum -------------------
__global__ void vq_gather_loss_kernel(const float* __restrict__ x,
                                      const float* __restrict__ wt,
                                      const int* __restrict__ idx,
                                      float* __restrict__ outq,
                                      float* __restrict__ lacc) {
  int row = blockIdx.x;
  int d = threadIdx.x;
  int k = idx[row];
  float q = wt[(size_t)k * kD + d];
  float xv = x[(size_t)row * kD + d];
  outq[(size_t)row * kD + d] = q;    // quantized_st == quantized numerically
  float diff = q - xv;
  __shared__ float red[256];
  red[d] = diff * diff;
  __syncthreads();
  for (int s = 128; s > 0; s >>= 1) {
    if (d < s) red[d] += red[d + s];
    __syncthreads();
  }
  if (d == 0) atomicAdd(lacc, red[0]);
}

// --- stream the 1.07 GB one-hot encodings (nontemporal, 8B stores since the
// encodings offset within d_out is only 8-byte aligned) ----------------------
__global__ void vq_encodings_kernel(const int* __restrict__ idx,
                                    float* __restrict__ enc) {
  size_t t = (size_t)blockIdx.x * 256 + threadIdx.x;
  size_t e = t * 2;
  int n = (int)(e >> 13);            // / kK
  int kb = (int)(e & (kK - 1));
  int k = idx[n];                    // broadcast within each row's 4096 threads
  v2f v;
  v.x = (k == kb) ? 1.f : 0.f;
  v.y = (k == kb + 1) ? 1.f : 0.f;
  __builtin_nontemporal_store(v, reinterpret_cast<v2f*>(enc) + t);
}

// --- loss + perplexity finalizer -------------------------------------------
__global__ void vq_finalize_kernel(const unsigned int* __restrict__ cnt,
                                   const float* __restrict__ lacc,
                                   float* __restrict__ out) {
  __shared__ float red[256];
  float s = 0.f;
  for (int k = threadIdx.x; k < kK; k += 256) {
    float p = (float)cnt[k] * (1.f / (float)kN);
    s += p * logf(p + 1e-10f);
  }
  red[threadIdx.x] = s;
  __syncthreads();
  for (int t = 128; t > 0; t >>= 1) {
    if (threadIdx.x < t) red[threadIdx.x] += red[threadIdx.x + t];
    __syncthreads();
  }
  if (threadIdx.x == 0) {
    out[OUT_LOSS] = lacc[0] * 1.25f / (float)((size_t)kN * kD);
    out[OUT_PPL] = expf(-red[0]);
  }
}

extern "C" void kernel_launch(void* const* d_in, const int* in_sizes, int n_in,
                              void* d_out, int out_size, void* d_ws, size_t ws_size,
                              hipStream_t stream) {
  (void)in_sizes; (void)n_in; (void)out_size; (void)ws_size;
  const float* x = (const float*)d_in[0];   // [N, D]
  const float* w = (const float*)d_in[1];   // [D, K]
  float* out = (float*)d_out;
  char* ws = (char*)d_ws;                   // needs ~51 MB

  unsigned short* whi = (unsigned short*)(ws + OFF_WHI);
  unsigned short* wlo = (unsigned short*)(ws + OFF_WLO);
  unsigned short* xhi = (unsigned short*)(ws + OFF_XHI);
  unsigned short* xlo = (unsigned short*)(ws + OFF_XLO);
  float* wt   = (float*)(ws + OFF_WT);
  float* wsq  = (float*)(ws + OFF_WSQ);
  float* xsq  = (float*)(ws + OFF_XSQ);
  int*   idx  = (int*)(ws + OFF_IDX);
  unsigned int* cnt = (unsigned int*)(ws + OFF_CNT);
  float* lacc = (float*)(ws + OFF_LACC);

  vq_zero_kernel<<<(kK + 256) / 256, 256, 0, stream>>>(cnt);
  vq_xsq_kernel<<<kN / 256, 256, 0, stream>>>(x, xsq);
  vq_wsq_wt_kernel<<<kK / 256, 256, 0, stream>>>(w, wsq, wt);
  vq_packx_kernel<<<kNRT * kNDT / 8, 256, 0, stream>>>(x, xhi, xlo);
  vq_packw_kernel<<<kNKT * kNDT / 8, 256, 0, stream>>>(w, whi, wlo);
  vq_gemm_argmin_kernel<<<kNRT / 4, 128, 0, stream>>>(xhi, xlo, whi, wlo, xsq,
                                                      wsq, idx, cnt,
                                                      out + OUT_IDX);
  vq_gather_loss_kernel<<<kN, 256, 0, stream>>>(x, wt, idx, out + OUT_Q, lacc);
  vq_encodings_kernel<<<(unsigned)((size_t)kN * kK / 512), 256, 0, stream>>>(
      idx, out + OUT_ENC);
  vq_finalize_kernel<<<1, 256, 0, stream>>>(cnt, lacc, out);
}
